// MLP_89163521065177
// MI455X (gfx1250) — matrix operand order repaired and checked by reading, not compiled, not run on hardware
//
#include <hip/hip_runtime.h>
#include <hip/hip_bf16.h>

// ---------------------------------------------------------------------------
// out = LeakyReLU( segsum(msg,index) @ (W2@W1)^T + count*(W2@b1) + b2 )
// (linear-through-linear collapse of the reference; scatter dominates: 410 MB)
// ---------------------------------------------------------------------------

typedef __attribute__((ext_vector_type(16))) __bf16 v16bf;
typedef __attribute__((ext_vector_type(8)))  __bf16 v8bf;
typedef __attribute__((ext_vector_type(8)))  float  v8f;
typedef __attribute__((ext_vector_type(4)))  float  f32x4;

#define OUTD 128
#define HIDD 512
#define TILE_M 32

// --- 1) zero agg+counts (N*129 floats) -------------------------------------
__global__ __launch_bounds__(256) void zero_kernel(float* __restrict__ p, size_t n) {
    size_t i = (size_t)blockIdx.x * blockDim.x + threadIdx.x;
    size_t stride = (size_t)gridDim.x * blockDim.x;
    for (; i < n; i += stride) p[i] = 0.0f;
}

// --- 2) fold weights: Wc = W2@W1 (bf16, row-major [n][k]), bc = W2@b1 (f32) -
__global__ __launch_bounds__(256) void fold_weights_kernel(
    const float* __restrict__ W1, const float* __restrict__ b1,
    const float* __restrict__ W2, __bf16* __restrict__ Wcbf,
    float* __restrict__ bc) {
    int t = blockIdx.x * blockDim.x + threadIdx.x;
    if (t < OUTD * OUTD) {
        int i = t >> 7;        // output row n
        int j = t & 127;       // input col k
        float acc = 0.0f;
        for (int k = 0; k < HIDD; ++k)
            acc += W2[i * HIDD + k] * W1[k * OUTD + j];
        Wcbf[i * OUTD + j] = (__bf16)acc;
    } else if (t < OUTD * OUTD + OUTD) {
        int i = t - OUTD * OUTD;
        float acc = 0.0f;
        for (int k = 0; k < HIDD; ++k)
            acc += W2[i * HIDD + k] * b1[k];
        bc[i] = acc;
    }
}

// --- 3) scatter: one wave per edge row; f32 HW atomics into L2-resident agg -
__global__ __launch_bounds__(256) void scatter_kernel(
    const float* __restrict__ msg, const int* __restrict__ index,
    float* __restrict__ agg, float* __restrict__ counts, int E) {
    int tid  = blockIdx.x * blockDim.x + threadIdx.x;
    int wave = tid >> 5;
    int lane = tid & 31;
    int nwaves = (gridDim.x * blockDim.x) >> 5;
    for (int e = wave; e < E; e += nwaves) {
        int idx = index[e];
        // 32 lanes x float4 = full 512B row, coalesced; NT: stream, don't pollute L2
        f32x4 v = __builtin_nontemporal_load(
            (const f32x4*)(msg + (size_t)e * OUTD + lane * 4));
        float* dst = agg + (size_t)idx * OUTD + lane * 4;
        unsafeAtomicAdd(dst + 0, v.x);
        unsafeAtomicAdd(dst + 1, v.y);
        unsafeAtomicAdd(dst + 2, v.z);
        unsafeAtomicAdd(dst + 3, v.w);
        if (lane == 0) unsafeAtomicAdd(counts + idx, 1.0f);
    }
}

__device__ __forceinline__ float leaky(float x) {
    return (x >= 0.0f) ? x : 0.01f * x;
}

// --- 4) fused GEMM: out = leaky(aggBf16 @ Wc^T + cnt*bc + b2), bf16 WMMA ----
__global__ __launch_bounds__(256) void gemm_kernel(
    const float* __restrict__ agg, const float* __restrict__ cnt,
    const __bf16* __restrict__ Wcbf, const float* __restrict__ bc,
    const float* __restrict__ b2, float* __restrict__ out, int N) {
    // row stride 144 bf16 = 72 words -> 8 mod 64: optimal LDS bank spread
    __shared__ __align__(16) __bf16 sA[TILE_M][144];
    __shared__ float sCnt[TILE_M];

    const int tid   = threadIdx.x;
    const int node0 = blockIdx.x * TILE_M;

    // cooperatively stage the 32x128 f32 agg tile into LDS as bf16
    for (int idx = tid; idx < TILE_M * OUTD; idx += 256) {
        int m = idx >> 7, k = idx & 127;
        int node = node0 + m;
        float v = (node < N) ? agg[(size_t)node * OUTD + k] : 0.0f;
        sA[m][k] = (__bf16)v;
    }
    if (tid < TILE_M) {
        int node = node0 + tid;
        sCnt[tid] = (node < N) ? cnt[node] : 0.0f;
    }
    __syncthreads();

    const int wave = tid >> 5;      // 8 waves -> 8 column tiles of 16
    const int lane = tid & 31;
    const int kgrp = lane >> 4;     // 0 or 1 (lane half)
    const int nlo  = lane & 15;
    const int n    = wave * 16 + nlo;

    // C init: bias = cnt[m]*bc[n] + b2[n]; C layout VGPR r <-> M=r+8*kgrp
    const float bcn = bc[n];
    const float b2n = b2[n];
    v8f c0, c1;
#pragma unroll
    for (int r = 0; r < 8; ++r) {
        int mrow = r + kgrp * 8;
        c0[r] = sCnt[mrow]      * bcn + b2n;
        c1[r] = sCnt[16 + mrow] * bcn + b2n;
    }

    const int mA0 = nlo;            // A fragment rows: lane&15 for both halves
    const int mA1 = 16 + nlo;
#pragma unroll
    for (int kk = 0; kk < OUTD; kk += 32) {
        // A 16x32 bf16: elems 0..7 = K kbase..+7, elems 8..15 = K kbase+16..+23
        int ka = kk + kgrp * 8;
        v8bf a0lo = *(const v8bf*)&sA[mA0][ka];
        v8bf a0hi = *(const v8bf*)&sA[mA0][ka + 16];
        v8bf a1lo = *(const v8bf*)&sA[mA1][ka];
        v8bf a1hi = *(const v8bf*)&sA[mA1][ka + 16];
        // B 32x16 bf16: lane holds column n; elems 0..7 = K kbase..+7, elems 8..15 = K kbase+16..+23
        int kb = kk + kgrp * 8;
        const __bf16* bp = Wcbf + (size_t)n * OUTD + kb;
        v8bf blo = *(const v8bf*)(bp);
        v8bf bhi = *(const v8bf*)(bp + 16);
        v16bf a0 = __builtin_shufflevector(a0lo, a0hi, 0,1,2,3,4,5,6,7,8,9,10,11,12,13,14,15);
        v16bf a1 = __builtin_shufflevector(a1lo, a1hi, 0,1,2,3,4,5,6,7,8,9,10,11,12,13,14,15);
        v16bf b  = __builtin_shufflevector(blo,  bhi,  0,1,2,3,4,5,6,7,8,9,10,11,12,13,14,15);
        c0 = __builtin_amdgcn_wmma_f32_16x16x32_bf16(false, a0, false, b, (short)0, c0, false, false);
        c1 = __builtin_amdgcn_wmma_f32_16x16x32_bf16(false, a1, false, b, (short)0, c1, false, false);
    }

    // Epilogue: block-uniform branch keeps the common path branch-free so the
    // 16 global_store_b32 can issue as clauses (only the last block is partial).
    float* outp = out + (size_t)(node0) * OUTD + n;
    if (node0 + TILE_M <= N) {
#pragma unroll
        for (int r = 0; r < 8; ++r) {
            int mrow = r + kgrp * 8;
            outp[(size_t)mrow * OUTD]        = leaky(c0[r]);
            outp[(size_t)(16 + mrow) * OUTD] = leaky(c1[r]);
        }
    } else {
#pragma unroll
        for (int r = 0; r < 8; ++r) {
            int mrow = r + kgrp * 8;
            if (node0 + mrow < N)      outp[(size_t)mrow * OUTD]        = leaky(c0[r]);
            if (node0 + 16 + mrow < N) outp[(size_t)(16 + mrow) * OUTD] = leaky(c1[r]);
        }
    }
}

// ---------------------------------------------------------------------------
extern "C" void kernel_launch(void* const* d_in, const int* in_sizes, int n_in,
                              void* d_out, int out_size, void* d_ws, size_t ws_size,
                              hipStream_t stream) {
    // setup_inputs order: msg, x_i, x_j, e_ij, index, num_nodes, W1, b1, W2, b2
    const float* msg   = (const float*)d_in[0];
    const int*   index = (const int*)d_in[4];
    const float* W1    = (const float*)d_in[6];
    const float* b1    = (const float*)d_in[7];
    const float* W2    = (const float*)d_in[8];
    const float* b2    = (const float*)d_in[9];

    const int E = in_sizes[4];
    const int N = out_size / OUTD;

    // workspace carve (~25.9 MB): agg | counts | Wcbf | bc
    float*  agg    = (float*)d_ws;
    float*  counts = agg + (size_t)N * OUTD;
    __bf16* Wcbf   = (__bf16*)(counts + N);        // byte off N*129*4: 16B aligned
    float*  bc     = (float*)(Wcbf + OUTD * OUTD);

    zero_kernel<<<1024, 256, 0, stream>>>(agg, (size_t)N * (OUTD + 1));
    fold_weights_kernel<<<(OUTD * OUTD + OUTD + 255) / 256, 256, 0, stream>>>(
        W1, b1, W2, Wcbf, bc);
    scatter_kernel<<<2048, 256, 0, stream>>>(msg, index, agg, counts, E);
    gemm_kernel<<<(N + TILE_M - 1) / TILE_M, 256, 0, stream>>>(
        agg, counts, Wcbf, bc, b2, (float*)d_out, N);
}
